// BigramLanguageModel_3762391351427
// MI455X (gfx1250) — matrix operand order repaired and checked by reading, not compiled, not run on hardware
//
#include <hip/hip_runtime.h>
#include <hip/hip_bf16.h>

typedef __attribute__((ext_vector_type(16))) _Float16 v16h;
typedef __attribute__((ext_vector_type(8)))  float    v8f;
typedef __attribute__((ext_vector_type(4)))  int      v4i;

#define VOCAB   256
#define CEMB    32
#define SEQT    8
#define NHEAD   4
#define HEADD   8
#define NLAYER  3
#define NBATCH  32768
#define NTOK    (NBATCH * SEQT)     // 262144 tokens
#define TILE    16                  // tokens per wave tile (2 sequences)
#define NTILES  (NTOK / TILE)       // 16384
#define WAVES   8                   // waves per workgroup
#define NWG     (NTILES / WAVES)    // 2048 workgroups
#define NWTILE  40                  // 24 layer B-tiles + 16 LM-head B-tiles
#define IMG_BYTES (NWTILE * 512 * 2)       // 40960 B of pre-swizzled f16 images
#define WS_IMG_OFF 8192                    // d_ws: [0,8K) partials, [8K,..) images
#define ATT_SCALE 0.17677669529663687f     // N_EMBD^-0.5 (reference scales by C)

#if __has_builtin(__builtin_amdgcn_global_load_async_to_lds_b128) && \
    __has_builtin(__builtin_amdgcn_s_wait_asynccnt)
#define HAVE_GFX1250_ASYNC 1
#endif

// Wave-private LDS staging only needs intra-wave ordering: CDNA5 LDS ops from a
// single wave execute in order, so a compiler code-motion barrier suffices.
__device__ __forceinline__ void wave_sync() {
    __builtin_amdgcn_fence(__ATOMIC_ACQ_REL, "wavefront");
    __builtin_amdgcn_wave_barrier();
    __builtin_amdgcn_fence(__ATOMIC_ACQ_REL, "wavefront");
}

// ---------------------------------------------------------------------------
// One-WG pre-pass: scatter-gather the fp32 weights once and emit the WMMA
// B-operand images (lane-major: image[tile][lane][half], 32B per lane) so the
// 2048 hot workgroups only do a contiguous 40KB global->LDS block copy.
// Tile order: t = layer*8 + kind*2 + nchunk (kind: q,k,v,ff), then 16 Wlm tiles.
// B-operand k order chosen here (A in buildA uses the same): n = lane&15, K = (lane&16) + half_index.
// ---------------------------------------------------------------------------
__global__ void swizzle_weights(const float* __restrict__ Wq, const float* __restrict__ Wk,
                                const float* __restrict__ Wv, const float* __restrict__ Wff,
                                const float* __restrict__ Wlm, _Float16* __restrict__ gimg) {
    const int tid = threadIdx.x;
    for (int e = tid; e < 24 * 512; e += 256) {
        int t = e >> 9, r = e & 511, l2 = r >> 4, i = r & 15;
        int k = (l2 & 16) + i, n16 = l2 & 15;
        int layer = t >> 3, j = t & 7, kind = j >> 1, c = j & 1;
        int n = c * 16 + n16;
        float val;
        if (kind < 3) {
            int h = n >> 3, d = n & 7;
            const float* W = (kind == 0) ? Wq : (kind == 1) ? Wk : Wv;
            val = W[((layer * NHEAD + h) * CEMB + k) * HEADD + d];
        } else {
            val = Wff[(layer * CEMB + k) * CEMB + n];
        }
        gimg[e] = (_Float16)val;
    }
    for (int e = tid; e < 16 * 512; e += 256) {
        int c = e >> 9, r = e & 511, l2 = r >> 4, i = r & 15;
        int k = (l2 & 16) + i, n = c * 16 + (l2 & 15);
        gimg[24 * 512 + e] = (_Float16)Wlm[k * VOCAB + n];
    }
}

__global__ __launch_bounds__(256)
void gpt_fused_fwd(const int* __restrict__ idx, const int* __restrict__ targets,
                   const float* __restrict__ tok_emb, const float* __restrict__ pos_emb,
                   const float* __restrict__ bff, const float* __restrict__ blm,
                   const _Float16* __restrict__ gimg,
                   float* __restrict__ out, float* __restrict__ nll_part) {
    // ---- LDS: pre-swizzled WMMA B-operand images (lane-major, v16h per lane) --
    __shared__ alignas(32) _Float16 wimg[NWTILE][32][16];
    __shared__ float ps[SEQT][CEMB];
    __shared__ float bffs[NLAYER][CEMB];
    __shared__ float blms[VOCAB];
    // ---- per-wave fp32 staging (D-layout round trips for attention) ----
    __shared__ float xs[WAVES][TILE][CEMB];
    __shared__ float qs[WAVES][TILE][CEMB];
    __shared__ float ks[WAVES][TILE][CEMB];
    __shared__ float vs[WAVES][TILE][CEMB];
    __shared__ float nllbuf[WAVES][TILE];

    const int tid  = threadIdx.x;
    const int lane = tid & 31;
    const int w    = tid >> 5;

    // Warm L2 with the token-embedding table (gathered by every tile)
    if (tid < 128) __builtin_prefetch((const char*)tok_emb + tid * 256, 0, 1);

    // ---- contiguous 40KB block copy of B-operand images: global -> LDS -------
#if defined(HAVE_GFX1250_ASYNC)
    {   // CDNA5 async DMA path: GLOBAL_LOAD_ASYNC_TO_LDS_B128 (ASYNCcnt)
        const char* gsrc = (const char*)gimg;
        char*       ldst = (char*)&wimg[0][0][0];
        #pragma unroll
        for (int jj = 0; jj < IMG_BYTES / (256 * 16); ++jj) {   // 10 iterations
            int off = (tid + jj * 256) * 16;
            __builtin_amdgcn_global_load_async_to_lds_b128(
                (__attribute__((address_space(1))) v4i*)(uintptr_t)(gsrc + off),
                (__attribute__((address_space(3))) v4i*)(unsigned)(uintptr_t)(ldst + off),
                0, 0);
        }
        __builtin_amdgcn_s_wait_asynccnt(0);
    }
#else
    {   // fallback: vector loads through VGPRs
        const float4* s4 = (const float4*)gimg;
        float4*       d4 = (float4*)&wimg[0][0][0];
        for (int e = tid; e < IMG_BYTES / 16; e += 256) d4[e] = s4[e];
    }
#endif
    for (int e = tid; e < SEQT * CEMB; e += 256) ps[e / CEMB][e % CEMB] = pos_emb[e];
    for (int e = tid; e < NLAYER * CEMB; e += 256) bffs[e / CEMB][e % CEMB] = bff[e];
    for (int e = tid; e < VOCAB; e += 256) blms[e] = blm[e];
    __syncthreads();   // cross-wave: weights/biases visible to all waves

    const int tile_id    = blockIdx.x * WAVES + w;
    const int token_base = tile_id * TILE;     // tile aligned to 2 full sequences
    const int nidx = lane & 15;                // D/B column within 16-chunk
    const int roff = (lane >> 4) << 3;         // D rows: 0..7 or 8..15
    const int kadd = (lane >> 4) << 4;         // A K-offset per half-wave

    // Build f16 A operand from xs with exact power-of-2 normalization.
    // The A layout partitions the 16x32 tile exactly across the 32 lanes, so a
    // register pass + full-wave max reduction sees every element once.
    auto buildA = [&](float& inv_scale) -> v16h {
        float xv[16];
        #pragma unroll
        for (int i = 0; i < 16; ++i) {
            int kk = kadd + i;
            xv[i] = xs[w][lane & 15][kk];
        }
        float mx = 0.f;
        #pragma unroll
        for (int i = 0; i < 16; ++i) mx = fmaxf(mx, fabsf(xv[i]));
        #pragma unroll
        for (int mk = 1; mk < 32; mk <<= 1) mx = fmaxf(mx, __shfl_xor(mx, mk, 32));
        int ex;
        frexpf(mx, &ex);                        // mx == 0 -> ex == 0 -> scale 1
        float sc  = ldexpf(1.f, -ex);           // x*sc in [0.5, 1): full f16 mantissa
        inv_scale = ldexpf(1.f,  ex);           // exact compensation on fp32 D
        v16h a;
        #pragma unroll
        for (int i = 0; i < 16; ++i) a[i] = (_Float16)(xv[i] * sc);
        return a;
    };

    // ================= x0 = tok_emb[idx] + pos_emb =============================
    for (int f = lane; f < TILE * CEMB; f += 32) {
        int row = f >> 5, col = f & 31;
        int tok = token_base + row;
        xs[w][row][col] = tok_emb[idx[tok] * CEMB + col] + ps[tok & 7][col];
    }
    wave_sync();

    v8f zc = {};
    // ================= transformer layers ======================================
    for (int l = 0; l < NLAYER; ++l) {
        float isc;
        v16h a = buildA(isc);
        // QKV: 6 WMMAs (2 N-chunks each), spill D to LDS in natural layout
        #pragma unroll
        for (int c = 0; c < 2; ++c) {
            v16h bq = *(const v16h*)&wimg[l * 8 + 0 + c][lane][0];
            v16h bk = *(const v16h*)&wimg[l * 8 + 2 + c][lane][0];
            v16h bv = *(const v16h*)&wimg[l * 8 + 4 + c][lane][0];
            v8f dq = __builtin_amdgcn_wmma_f32_16x16x32_f16(false, a, false, bq, (short)0, zc, false, false);
            v8f dk = __builtin_amdgcn_wmma_f32_16x16x32_f16(false, a, false, bk, (short)0, zc, false, false);
            v8f dv = __builtin_amdgcn_wmma_f32_16x16x32_f16(false, a, false, bv, (short)0, zc, false, false);
            #pragma unroll
            for (int r = 0; r < 8; ++r) {
                int rr = roff + r, cc = c * 16 + nidx;
                qs[w][rr][cc] = dq[r] * isc;
                ks[w][rr][cc] = dk[r] * isc;
                vs[w][rr][cc] = dv[r] * isc;
            }
        }
        wave_sync();
        // --------- attention: 64 (token,head) pairs -> 2 per lane -------------
        #pragma unroll
        for (int p = 0; p < 2; ++p) {
            int pid  = lane + p * 32;
            int tok  = pid >> 2, h = pid & 3;
            int t_in = tok & 7, sb = tok & ~7, hb = h * HEADD;
            float qv[8];
            #pragma unroll
            for (int d = 0; d < 8; ++d) qv[d] = qs[w][tok][hb + d];
            float wei[8], m = -1e30f;
            #pragma unroll
            for (int s = 0; s < 8; ++s) {
                float acc = 0.f;
                #pragma unroll
                for (int d = 0; d < 8; ++d) acc += qv[d] * ks[w][sb + s][hb + d];
                acc = (s <= t_in) ? acc * ATT_SCALE : -1e30f;   // causal mask
                wei[s] = acc; m = fmaxf(m, acc);
            }
            float sum = 0.f;
            #pragma unroll
            for (int s = 0; s < 8; ++s) {
                float e = (s <= t_in) ? __expf(wei[s] - m) : 0.f;
                wei[s] = e; sum += e;
            }
            float inv = 1.f / sum;
            float o[8];
            #pragma unroll
            for (int d = 0; d < 8; ++d) o[d] = 0.f;
            #pragma unroll
            for (int s = 0; s < 8; ++s) {
                float wn = wei[s] * inv;
                #pragma unroll
                for (int d = 0; d < 8; ++d) o[d] += wn * vs[w][sb + s][hb + d];
            }
            #pragma unroll
            for (int d = 0; d < 8; ++d) xs[w][tok][hb + d] = o[d];   // concat heads
        }
        wave_sync();
        // --------- FFN: relu(x @ Wff + bff) -----------------------------------
        float isc2;
        v16h a2 = buildA(isc2);
        wave_sync();   // xs reads complete before overwrite (WAR)
        #pragma unroll
        for (int c = 0; c < 2; ++c) {
            v16h bf = *(const v16h*)&wimg[l * 8 + 6 + c][lane][0];
            v8f df = __builtin_amdgcn_wmma_f32_16x16x32_f16(false, a2, false, bf, (short)0, zc, false, false);
            #pragma unroll
            for (int r = 0; r < 8; ++r) {
                float v = df[r] * isc2 + bffs[l][c * 16 + nidx];
                xs[w][roff + r][c * 16 + nidx] = fmaxf(v, 0.f);
            }
        }
        wave_sync();
    }

    // ================= LM head + streamed log-softmax ==========================
    float isc3;
    v16h a3 = buildA(isc3);
    int tgt[8];
    float mrow[8], srow[8], tlog[8];
    #pragma unroll
    for (int r = 0; r < 8; ++r) {
        tgt[r] = targets[token_base + roff + r];
        mrow[r] = -1e30f; srow[r] = 0.f; tlog[r] = 0.f;
    }
    for (int c = 0; c < 16; ++c) {          // 16 WMMAs over vocab
        v16h bl = *(const v16h*)&wimg[24 + c][lane][0];
        v8f dl = __builtin_amdgcn_wmma_f32_16x16x32_f16(false, a3, false, bl, (short)0, zc, false, false);
        int col = c * 16 + nidx;
        float bb = blms[col];
        #pragma unroll
        for (int r = 0; r < 8; ++r) {
            float val = dl[r] * isc3 + bb;
            // streaming 256MB: bypass L2 (write-once, never re-read; L2 is 192MB)
            __builtin_nontemporal_store(val,
                out + (size_t)(token_base + roff + r) * VOCAB + col);
            float mn = fmaxf(mrow[r], val);                     // online logsumexp
            srow[r] = srow[r] * __expf(mrow[r] - mn) + __expf(val - mn);
            mrow[r] = mn;
            tlog[r] += (col == tgt[r]) ? val : 0.f;             // capture target logit
        }
    }
    // reduce over the 16 lanes holding the same rows (xor masks stay in half-wave)
    #pragma unroll
    for (int mask = 1; mask <= 8; mask <<= 1) {
        #pragma unroll
        for (int r = 0; r < 8; ++r) {
            float mo = __shfl_xor(mrow[r], mask, 32);
            float so = __shfl_xor(srow[r], mask, 32);
            float to = __shfl_xor(tlog[r], mask, 32);
            float mn = fmaxf(mrow[r], mo);
            srow[r] = srow[r] * __expf(mrow[r] - mn) + so * __expf(mo - mn);
            mrow[r] = mn;
            tlog[r] += to;
        }
    }
    if ((lane & 15) == 0) {
        #pragma unroll
        for (int r = 0; r < 8; ++r)
            nllbuf[w][roff + r] = (mrow[r] + __logf(srow[r])) - tlog[r];
    }
    __syncthreads();   // cross-wave: nllbuf
    if (tid == 0) {    // deterministic fixed-order per-WG partial sum
        float acc = 0.f;
        for (int w2 = 0; w2 < WAVES; ++w2)
            for (int r = 0; r < TILE; ++r) acc += nllbuf[w2][r];
        nll_part[blockIdx.x] = acc;
    }
}

__global__ void loss_reduce(const float* __restrict__ part, float* __restrict__ out) {
    if (threadIdx.x == 0 && blockIdx.x == 0) {
        float acc = 0.f;
        for (int i = 0; i < NWG; ++i) acc += part[i];   // deterministic order
        out[(size_t)NTOK * VOCAB] = acc / (float)NTOK;
    }
}

extern "C" void kernel_launch(void* const* d_in, const int* in_sizes, int n_in,
                              void* d_out, int out_size, void* d_ws, size_t ws_size,
                              hipStream_t stream) {
    const int*   idx     = (const int*)d_in[0];
    const int*   targets = (const int*)d_in[1];
    const float* tok_emb = (const float*)d_in[2];
    const float* pos_emb = (const float*)d_in[3];
    const float* Wq      = (const float*)d_in[4];
    const float* Wk      = (const float*)d_in[5];
    const float* Wv      = (const float*)d_in[6];
    const float* Wff     = (const float*)d_in[7];
    const float* bff     = (const float*)d_in[8];
    const float* Wlm     = (const float*)d_in[9];
    const float* blm     = (const float*)d_in[10];
    float* out  = (float*)d_out;
    // d_ws layout: [0, 8KB) per-WG NLL partials; [8KB, 48KB) swizzled f16 images
    float*    part = (float*)d_ws;
    _Float16* gimg = (_Float16*)((char*)d_ws + WS_IMG_OFF);

    swizzle_weights<<<dim3(1), dim3(256), 0, stream>>>(Wq, Wk, Wv, Wff, Wlm, gimg);
    gpt_fused_fwd<<<dim3(NWG), dim3(256), 0, stream>>>(
        idx, targets, tok_emb, pos_emb, bff, blm, gimg, out, part);
    loss_reduce<<<dim3(1), dim3(32), 0, stream>>>(part, out);
}
